// MLAAttention_44203803410465
// MI455X (gfx1250) — compile-verified, matrix operand order checked by reading, not run on hardware
//
#include <hip/hip_runtime.h>
#include <hip/hip_bf16.h>

// ---------------- problem constants ----------------
#define S_LEN 2048
#define H_DIM 2048
#define NHEAD 32
#define QLR   1344
#define KVLR  128
#define DNOPE 32
#define DROPE 32
#define DVDIM 64

// ---------------- types ----------------
typedef __attribute__((ext_vector_type(16))) __bf16 bf16x16;
typedef __attribute__((ext_vector_type(8)))  __bf16 bf16x8;
typedef __attribute__((ext_vector_type(8)))  float  f32x8;

union frag16 { bf16x16 v; bf16x8 h[2]; };

__device__ __forceinline__ __bf16 f2bf(float f) {
    unsigned u = __builtin_bit_cast(unsigned, f);
    unsigned r = u + 0x7FFFu + ((u >> 16) & 1u);   // round-to-nearest-even
    unsigned short hs = (unsigned short)(r >> 16);
    return __builtin_bit_cast(__bf16, hs);
}

__device__ __forceinline__ f32x8 zero8() {
    f32x8 z;
#pragma unroll
    for (int i = 0; i < 8; ++i) z[i] = 0.0f;
    return z;
}

// =====================================================================
// Elementwise fp32 -> bf16 (n must be a multiple of 4)
// =====================================================================
__global__ void __launch_bounds__(256)
convert_f32_bf16(const float* __restrict__ in, __bf16* __restrict__ out, int n)
{
    int idx = (blockIdx.x * 256 + threadIdx.x) * 4;
    for (; idx + 3 < n; idx += gridDim.x * 256 * 4) {
        float4 v = *(const float4*)(in + idx);
        out[idx + 0] = f2bf(v.x);
        out[idx + 1] = f2bf(v.y);
        out[idx + 2] = f2bf(v.z);
        out[idx + 3] = f2bf(v.w);
    }
}

// =====================================================================
// Transpose + convert: in fp32 [R][C] -> out bf16 [C][R].
// R must be a multiple of 32; C ragged OK. Block 256 threads, 32x32 tile.
// =====================================================================
__global__ void __launch_bounds__(256)
transpose_convert(const float* __restrict__ in, __bf16* __restrict__ out,
                  int R, int C)
{
    __shared__ float tile[32][33];
    const int rb = blockIdx.x * 32;
    const int cb = blockIdx.y * 32;
    const int tx = threadIdx.x & 31;
    const int ty = threadIdx.x >> 5;      // 0..7
#pragma unroll
    for (int i = ty; i < 32; i += 8) {
        int c = cb + tx;
        tile[i][tx] = (c < C) ? in[(size_t)(rb + i) * C + c] : 0.0f;
    }
    __syncthreads();
#pragma unroll
    for (int i = ty; i < 32; i += 8) {
        int c = cb + i;
        if (c < C) out[(size_t)c * R + rb + tx] = f2bf(tile[tx][i]);
    }
}

// =====================================================================
// GEMM: C[M,N] = A[M,K] * B[K,N], A bf16 [M][K], Bt bf16 [N][K] (pre-
// transposed), C fp32. LDS-free: fragments are contiguous 16B global
// loads (A-frag interleave: lanes0-15 K{kb..kb+7,kb+16..kb+23}).
// Block 128 thr (4 waves, 2x2), block tile 128x128, wave tile 64x64.
// Requires M % 128 == 0, K % 32 == 0; N ragged (guarded at store; OOB
// B fragments read row 0, their columns are never stored).
// =====================================================================
__global__ void __launch_bounds__(128)
gemm_bf16_wmma(const __bf16* __restrict__ A, const __bf16* __restrict__ Bt,
               float* __restrict__ C, int M, int N, int K)
{
    const int tid  = threadIdx.x;
    const int wave = tid >> 5;
    const int lane = tid & 31;
    const int half = lane >> 4;
    const int l16  = lane & 15;
    const int kb   = half * 8;

    const int bm  = blockIdx.y * 128 + (wave >> 1) * 64;
    const int bn0 = blockIdx.x * 128 + (wave & 1) * 64;

    f32x8 acc[4][4];
#pragma unroll
    for (int mi = 0; mi < 4; ++mi)
#pragma unroll
        for (int ni = 0; ni < 4; ++ni) acc[mi][ni] = zero8();

    const __bf16* arow[4];
    const __bf16* brow[4];
#pragma unroll
    for (int mi = 0; mi < 4; ++mi)
        arow[mi] = A + (size_t)(bm + mi * 16 + l16) * K;
#pragma unroll
    for (int ni = 0; ni < 4; ++ni) {
        int col = bn0 + ni * 16 + l16;
        brow[ni] = Bt + (size_t)(col < N ? col : 0) * K;
    }

    for (int k0 = 0; k0 < K; k0 += 32) {
        __builtin_prefetch(arow[0] + k0 + 128, 0, 1);
        __builtin_prefetch(brow[0] + k0 + 128, 0, 1);
        frag16 af[4], bf[4];
#pragma unroll
        for (int mi = 0; mi < 4; ++mi) {
            af[mi].h[0] = *(const bf16x8*)(arow[mi] + k0 + kb);
            af[mi].h[1] = *(const bf16x8*)(arow[mi] + k0 + kb + 16);
        }
#pragma unroll
        for (int ni = 0; ni < 4; ++ni) {
            bf[ni].h[0] = *(const bf16x8*)(brow[ni] + k0 + kb);
            bf[ni].h[1] = *(const bf16x8*)(brow[ni] + k0 + kb + 16);
        }
#pragma unroll
        for (int mi = 0; mi < 4; ++mi)
#pragma unroll
            for (int ni = 0; ni < 4; ++ni)
                acc[mi][ni] = __builtin_amdgcn_wmma_f32_16x16x32_bf16(
                    false, af[mi].v, false, bf[ni].v, (short)0, acc[mi][ni],
                    false, false);
    }

    // C-layout: VGPR i -> row i + 8*half, lane&15 -> col
#pragma unroll
    for (int ni = 0; ni < 4; ++ni) {
        int col = bn0 + ni * 16 + l16;
        if (col < N) {
#pragma unroll
            for (int mi = 0; mi < 4; ++mi)
#pragma unroll
                for (int i = 0; i < 8; ++i) {
                    int row = bm + mi * 16 + i + half * 8;
                    C[(size_t)row * N + col] = acc[mi][ni][i];
                }
        }
    }
}

// =====================================================================
// RMSNorm over rows, fp32 in -> bf16 out:
//   out[r,c] = bf16( w[c]*(x[r,c]*rsqrt(mean(x^2)+eps)) + b[c] )
// =====================================================================
__global__ void __launch_bounds__(256)
rmsnorm_kernel(const float* __restrict__ x, const float* __restrict__ w,
               const float* __restrict__ b, __bf16* __restrict__ out,
               int in_stride, int out_stride, int cols)
{
    __shared__ float red[256];
    const int row = blockIdx.x;
    const float* xr = x + (size_t)row * in_stride;
    __bf16* orow = out + (size_t)row * out_stride;

    float vals[6];
    float ss = 0.0f;
    int cnt = 0;
    for (int c = threadIdx.x; c < cols; c += 256) {
        float v = xr[c];
        vals[cnt++] = v;
        ss += v * v;
    }
    red[threadIdx.x] = ss;
    __syncthreads();
    for (int off = 128; off > 0; off >>= 1) {
        if ((int)threadIdx.x < off) red[threadIdx.x] += red[threadIdx.x + off];
        __syncthreads();
    }
    float inv = rsqrtf(red[0] / (float)cols + 1e-5f);
    cnt = 0;
    for (int c = threadIdx.x; c < cols; c += 256)
        orow[c] = f2bf(w[c] * (vals[cnt++] * inv) + b[c]);
}

// =====================================================================
// Assemble Q/K/V with YaRN RoPE (SCALING=1 -> inv_freq = BASE^(-j/16),
// mscale = 1). Folds 1/sqrt(64) into Q. Layouts:
//   Qh,Kh : [NH][S][64] bf16 ; Vt : [NH][64][S] bf16 (transposed)
// =====================================================================
__global__ void __launch_bounds__(256)
assemble_kernel(const float* __restrict__ q,    // [S][NH*64] fp32
                const float* __restrict__ kv,   // [S][160]   fp32
                const float* __restrict__ kvu,  // [S][NH*96] fp32
                __bf16* __restrict__ Qh, __bf16* __restrict__ Kh,
                __bf16* __restrict__ Vt)
{
    const int s = blockIdx.x;
    const float t = (float)s;
    for (int idx = threadIdx.x; idx < NHEAD * 64; idx += 256) {
        const int h = idx >> 6;
        const int d = idx & 63;
        float qv, kvl;
        if (d < DNOPE) {
            qv  = q[(size_t)s * (NHEAD * 64) + h * 64 + d];
            kvl = kvu[(size_t)s * (NHEAD * 96) + h * 96 + d];
        } else {
            const int j  = d - DNOPE;            // 0..31 rope output index
            const int jj = (j < 16) ? j : (j - 16);
            const float f = __powf(500000.0f, -(float)jj * (1.0f / 16.0f));
            float sn, c;
            __sincosf(t * f, &sn, &c);
            const float* qr = q + (size_t)s * (NHEAD * 64) + h * 64 + DNOPE;
            float e = qr[2 * jj], o = qr[2 * jj + 1];
            qv = (j < 16) ? (e * c - o * sn) : (o * c + e * sn);
            const float* kr = kv + (size_t)s * (KVLR + DROPE) + KVLR;
            float ke = kr[2 * jj], ko = kr[2 * jj + 1];
            kvl = (j < 16) ? (ke * c - ko * sn) : (ko * c + ke * sn);
        }
        Qh[((size_t)h * S_LEN + s) * 64 + d] = f2bf(qv * 0.125f);  // scale folded
        Kh[((size_t)h * S_LEN + s) * 64 + d] = f2bf(kvl);
        float vv = kvu[(size_t)s * (NHEAD * 96) + h * 96 + DNOPE + d];
        Vt[((size_t)h * 64 + d) * S_LEN + s] = f2bf(vv);
    }
}

// =====================================================================
// Flash attention, causal. Grid (S/64, NH), block 128 (4 waves).
// Each wave owns 16 query rows; online softmax over 64-key tiles.
// Emits ctx as bf16 (feeds the final GEMM directly).
// =====================================================================
__global__ void __launch_bounds__(128)
attn_kernel(const __bf16* __restrict__ Q,  // [NH][S][64]
            const __bf16* __restrict__ Km, // [NH][S][64]
            const __bf16* __restrict__ Vt, // [NH][64][S]
            __bf16* __restrict__ ctx)      // [S][NH*64]
{
    __shared__ __bf16 Plds[4][16][64];     // per-wave P strip

    const int h    = blockIdx.y;
    const int qt   = blockIdx.x;
    const int tid  = threadIdx.x;
    const int wave = tid >> 5;
    const int lane = tid & 31;
    const int half = lane >> 4;
    const int l16  = lane & 15;
    const int kb   = half * 8;

    const __bf16* Qh = Q  + (size_t)h * S_LEN * 64;
    const __bf16* Kh = Km + (size_t)h * S_LEN * 64;
    const __bf16* Vh = Vt + (size_t)h * 64 * S_LEN;

    frag16 qf[2];
    {
        const __bf16* qp = Qh + (size_t)(qt * 64 + wave * 16 + l16) * 64;
        qf[0].h[0] = *(const bf16x8*)(qp + kb);
        qf[0].h[1] = *(const bf16x8*)(qp + kb + 16);
        qf[1].h[0] = *(const bf16x8*)(qp + 32 + kb);
        qf[1].h[1] = *(const bf16x8*)(qp + 32 + kb + 16);
    }

    float m_i[8], l_i[8];
    f32x8 oacc[4];
#pragma unroll
    for (int i = 0; i < 8; ++i) { m_i[i] = -3.0e38f; l_i[i] = 0.0f; }
#pragma unroll
    for (int ns = 0; ns < 4; ++ns) oacc[ns] = zero8();

    for (int kt = 0; kt <= qt; ++kt) {
        // ---- scores: 16 rows x 64 keys ----
        f32x8 sacc[4];
#pragma unroll
        for (int ns = 0; ns < 4; ++ns) {
            sacc[ns] = zero8();
            const __bf16* kp = Kh + (size_t)(kt * 64 + ns * 16 + l16) * 64;
            frag16 b0, b1;
            b0.h[0] = *(const bf16x8*)(kp + kb);
            b0.h[1] = *(const bf16x8*)(kp + kb + 16);
            b1.h[0] = *(const bf16x8*)(kp + 32 + kb);
            b1.h[1] = *(const bf16x8*)(kp + 32 + kb + 16);
            sacc[ns] = __builtin_amdgcn_wmma_f32_16x16x32_bf16(
                false, qf[0].v, false, b0.v, (short)0, sacc[ns], false, false);
            sacc[ns] = __builtin_amdgcn_wmma_f32_16x16x32_bf16(
                false, qf[1].v, false, b1.v, (short)0, sacc[ns], false, false);
        }
        // ---- causal mask on diagonal tile ----
        if (kt == qt) {
#pragma unroll
            for (int ns = 0; ns < 4; ++ns) {
                int n = ns * 16 + l16;
#pragma unroll
                for (int i = 0; i < 8; ++i) {
                    int mrow = wave * 16 + i + half * 8;
                    if (n > mrow) sacc[ns][i] = -1.0e9f;
                }
            }
        }
        // ---- row max across 4 accs + 16 lanes of this half ----
        float rmax[8];
#pragma unroll
        for (int i = 0; i < 8; ++i) {
            float v = fmaxf(fmaxf(sacc[0][i], sacc[1][i]),
                            fmaxf(sacc[2][i], sacc[3][i]));
#pragma unroll
            for (int off = 1; off < 16; off <<= 1)
                v = fmaxf(v, __shfl_xor(v, off, 32));
            rmax[i] = v;
        }
        // ---- online softmax rescale ----
#pragma unroll
        for (int i = 0; i < 8; ++i) {
            float mnew  = fmaxf(m_i[i], rmax[i]);
            float scale = __expf(m_i[i] - mnew);
            m_i[i] = mnew;
            l_i[i] *= scale;
#pragma unroll
            for (int ns = 0; ns < 4; ++ns) oacc[ns][i] *= scale;
        }
        // ---- P = exp(s - m): stage to wave-private LDS ----
#pragma unroll
        for (int ns = 0; ns < 4; ++ns) {
#pragma unroll
            for (int i = 0; i < 8; ++i) {
                float p = __expf(sacc[ns][i] - m_i[i]);
                sacc[ns][i] = p;
                Plds[wave][i + half * 8][ns * 16 + l16] = f2bf(p);
            }
        }
        // ---- row sums ----
#pragma unroll
        for (int i = 0; i < 8; ++i) {
            float v = sacc[0][i] + sacc[1][i] + sacc[2][i] + sacc[3][i];
#pragma unroll
            for (int off = 1; off < 16; off <<= 1)
                v += __shfl_xor(v, off, 32);
            l_i[i] += v;
        }
        // ---- O += P @ V  (LDS ops in-order within a wave) ----
#pragma unroll
        for (int ks = 0; ks < 2; ++ks) {
            frag16 pf;
            const __bf16* pp = &Plds[wave][l16][ks * 32];
            pf.h[0] = *(const bf16x8*)(pp + kb);
            pf.h[1] = *(const bf16x8*)(pp + kb + 16);
#pragma unroll
            for (int ns = 0; ns < 4; ++ns) {
                const __bf16* vp =
                    Vh + (size_t)(ns * 16 + l16) * S_LEN + kt * 64 + ks * 32;
                frag16 bv;
                bv.h[0] = *(const bf16x8*)(vp + kb);
                bv.h[1] = *(const bf16x8*)(vp + kb + 16);
                oacc[ns] = __builtin_amdgcn_wmma_f32_16x16x32_bf16(
                    false, pf.v, false, bv.v, (short)0, oacc[ns], false, false);
            }
        }
    }

    // ---- finalize ----
#pragma unroll
    for (int ns = 0; ns < 4; ++ns) {
#pragma unroll
        for (int i = 0; i < 8; ++i) {
            int row = qt * 64 + wave * 16 + i + half * 8;
            int d   = ns * 16 + l16;
            ctx[(size_t)row * (NHEAD * 64) + h * 64 + d] =
                f2bf(oacc[ns][i] / l_i[i]);
        }
    }
}

// =====================================================================
// launch
// =====================================================================
extern "C" void kernel_launch(void* const* d_in, const int* in_sizes, int n_in,
                              void* d_out, int out_size, void* d_ws, size_t ws_size,
                              hipStream_t stream)
{
    (void)in_sizes; (void)n_in; (void)out_size; (void)ws_size;
    const float* x        = (const float*)d_in[0];
    const float* wq_down  = (const float*)d_in[1];
    const float* q_ln_w   = (const float*)d_in[2];
    const float* q_ln_b   = (const float*)d_in[3];
    const float* wq_up    = (const float*)d_in[4];
    const float* wkv_down = (const float*)d_in[5];
    const float* kv_ln_w  = (const float*)d_in[6];
    const float* kv_ln_b  = (const float*)d_in[7];
    const float* wkv_up   = (const float*)d_in[8];
    const float* wo       = (const float*)d_in[9];
    float* out = (float*)d_out;

    char* ws = (char*)d_ws;
    size_t off = 0;
    auto alloc = [&](size_t bytes) -> void* {
        void* p = ws + off;
        off += (bytes + 255) & ~(size_t)255;
        return p;
    };
    // bf16 operands
    __bf16* x_bf   = (__bf16*)alloc((size_t)S_LEN * H_DIM * 2);          // 8 MB
    __bf16* wqd_t  = (__bf16*)alloc((size_t)QLR * H_DIM * 2);            // 5.5 MB
    __bf16* wqu_t  = (__bf16*)alloc((size_t)(NHEAD * 64) * QLR * 2);     // 5.5 MB
    __bf16* wkvd_t = (__bf16*)alloc((size_t)(KVLR + DROPE) * H_DIM * 2); // 0.6 MB
    __bf16* wkvu_t = (__bf16*)alloc((size_t)(NHEAD * 96) * KVLR * 2);    // 0.8 MB
    __bf16* wo_t   = (__bf16*)alloc((size_t)H_DIM * (NHEAD * 64) * 2);   // 8 MB
    // activations
    float*  t_q    = (float*)alloc((size_t)S_LEN * QLR * 4);             // 11 MB
    __bf16* t_q_bf = (__bf16*)alloc((size_t)S_LEN * QLR * 2);            // 5.5 MB
    float*  qbuf   = (float*)alloc((size_t)S_LEN * NHEAD * 64 * 4);      // 16 MB
    float*  kvb    = (float*)alloc((size_t)S_LEN * (KVLR + DROPE) * 4);  // 1.3 MB
    __bf16* kvc_bf = (__bf16*)alloc((size_t)S_LEN * KVLR * 2);           // 0.5 MB
    float*  kvu    = (float*)alloc((size_t)S_LEN * NHEAD * 96 * 4);      // 24 MB
    __bf16* Qh     = (__bf16*)alloc((size_t)NHEAD * S_LEN * 64 * 2);     // 8 MB
    __bf16* Kh     = (__bf16*)alloc((size_t)NHEAD * S_LEN * 64 * 2);     // 8 MB
    __bf16* Vt     = (__bf16*)alloc((size_t)NHEAD * 64 * S_LEN * 2);     // 8 MB
    __bf16* ctx    = (__bf16*)alloc((size_t)S_LEN * NHEAD * 64 * 2);     // 8 MB

    const dim3 blk128(128), blk256(256);

    // ---- prep: convert x, transpose+convert weights ----
    convert_f32_bf16<<<4096, blk256, 0, stream>>>(x, x_bf, S_LEN * H_DIM);
    transpose_convert<<<dim3(H_DIM / 32, QLR / 32), blk256, 0, stream>>>(
        wq_down, wqd_t, H_DIM, QLR);
    transpose_convert<<<dim3(QLR / 32, (NHEAD * 64) / 32), blk256, 0, stream>>>(
        wq_up, wqu_t, QLR, NHEAD * 64);
    transpose_convert<<<dim3(H_DIM / 32, (KVLR + DROPE) / 32), blk256, 0, stream>>>(
        wkv_down, wkvd_t, H_DIM, KVLR + DROPE);
    transpose_convert<<<dim3(KVLR / 32, (NHEAD * 96) / 32), blk256, 0, stream>>>(
        wkv_up, wkvu_t, KVLR, NHEAD * 96);
    transpose_convert<<<dim3((NHEAD * 64) / 32, H_DIM / 32), blk256, 0, stream>>>(
        wo, wo_t, NHEAD * 64, H_DIM);

    // 1) q latent: x @ wq_down -> t_q  (M=2048, N=1344, K=2048)
    gemm_bf16_wmma<<<dim3((QLR + 127) / 128, S_LEN / 128), blk128, 0, stream>>>(
        x_bf, wqd_t, t_q, S_LEN, QLR, H_DIM);
    // 2) rmsnorm q latent -> bf16
    rmsnorm_kernel<<<S_LEN, blk256, 0, stream>>>(t_q, q_ln_w, q_ln_b, t_q_bf,
                                                 QLR, QLR, QLR);
    // 3) q up: t_q_bf @ wq_up -> qbuf  (N=2048, K=1344)
    gemm_bf16_wmma<<<dim3((NHEAD * 64) / 128, S_LEN / 128), blk128, 0, stream>>>(
        t_q_bf, wqu_t, qbuf, S_LEN, NHEAD * 64, QLR);
    // 4) kv latent: x @ wkv_down -> kvb  (N=160 ragged)
    gemm_bf16_wmma<<<dim3((KVLR + DROPE + 127) / 128, S_LEN / 128), blk128, 0, stream>>>(
        x_bf, wkvd_t, kvb, S_LEN, KVLR + DROPE, H_DIM);
    // 5) rmsnorm kv latent (first 128 cols) -> kvc_bf
    rmsnorm_kernel<<<S_LEN, blk256, 0, stream>>>(kvb, kv_ln_w, kv_ln_b, kvc_bf,
                                                 KVLR + DROPE, KVLR, KVLR);
    // 6) kv up: kvc_bf @ wkv_up -> kvu  (N=3072, K=128)
    gemm_bf16_wmma<<<dim3((NHEAD * 96) / 128, S_LEN / 128), blk128, 0, stream>>>(
        kvc_bf, wkvu_t, kvu, S_LEN, NHEAD * 96, KVLR);
    // 7) RoPE + assemble bf16 Q/K/V
    assemble_kernel<<<S_LEN, blk256, 0, stream>>>(qbuf, kvb, kvu, Qh, Kh, Vt);
    // 8) causal flash attention -> ctx (bf16)
    attn_kernel<<<dim3(S_LEN / 64, NHEAD), blk128, 0, stream>>>(Qh, Kh, Vt, ctx);
    // 9) output proj: ctx @ wo -> out  (fp32)
    gemm_bf16_wmma<<<dim3(H_DIM / 128, S_LEN / 128), blk128, 0, stream>>>(
        ctx, wo_t, out, S_LEN, H_DIM, NHEAD * 64);
}
